// DualAxisBlock_36352603193889
// MI455X (gfx1250) — matrix operand order repaired and checked by reading, not compiled, not run on hardware
//
#include <hip/hip_runtime.h>
#include <cstdint>

#define HEADS   8
#define HDIM    32
#define DMODEL  256
#define FFDIM   1024
#define TT      256   // time length
#define NST     256   // stocks
#define BB      2
#define TP      46    // active time steps: 30 strided (every 8) + 16 dense tail
#define NSPARSE 30
#define S1 (BB*NST*TT)   // 131072 tokens
#define S2 (BB*TP*NST)   // 23552 tokens
#define SCALE_QK 0.17677669529663687f

typedef __attribute__((ext_vector_type(16))) __bf16 v16bf;
typedef __attribute__((ext_vector_type(8)))  float  v8f;

union FragA { v16bf v; uint32_t u[8]; };
union FragC { v8f   v; float    f[8]; };

__device__ __forceinline__ uint16_t f2bf(float f){
  uint32_t u = __float_as_uint(f);
  u += 0x7fffu + ((u >> 16) & 1u);           // round-to-nearest-even
  return (uint16_t)(u >> 16);
}

__device__ __forceinline__ int act_idx(int i){
  return (i < NSPARSE) ? (i * 8) : (TT - 16 + (i - NSPARSE));
}

// tanh-approx GELU with exp-based tanh (one v_exp_f32, saturates correctly)
__device__ __forceinline__ float gelu_tanh(float x){
  float u = 0.79788456f * (x + 0.044715f * x * x * x);
  float t = 1.0f - 2.0f / (1.0f + __expf(2.0f * u));
  return 0.5f * x * (1.0f + t);
}

// CDNA5 async copy: global memory -> LDS, 16 bytes per lane, tracked by ASYNCcnt.
// vdst = LDS byte offset (flat LDS addr truncates to offset), vaddr = 64-bit global.
__device__ __forceinline__ void async_copy_b128(uint32_t lds_off, const void* gaddr){
  asm volatile("global_load_async_to_lds_b128 %0, %1, off"
               :: "v"(lds_off), "v"(gaddr)
               : "memory");
}
__device__ __forceinline__ void wait_async0(){
  asm volatile("s_wait_asynccnt 0x0" ::: "memory");
}
__device__ __forceinline__ uint32_t lds_off_of(const void* p){
  return (uint32_t)(uintptr_t)p;   // LDS aperture: addr[31:0] == LDS byte offset
}

// ---------------------------------------------------------------- utilities
// cast with transpose: src is K x N (f32), dst is N x K (bf16) so GEMM B tiles
// become contiguous along K and can be staged with async B128 copies.
__global__ void cast_transpose_bf16(const float* __restrict__ src,
                                    uint16_t* __restrict__ dst, int K, int N){
  int idx = blockIdx.x * 256 + threadIdx.x;
  if (idx >= K * N) return;
  int k = idx % K, n = idx / K;
  dst[(size_t)n * K + k] = f2bf(src[(size_t)k * N + n]);
}

__global__ void copy_f32v4(const float4* __restrict__ src,
                           float4* __restrict__ dst, int n4){
  int i = blockIdx.x * 256 + threadIdx.x;
  if (i < n4) dst[i] = src[i];
}

// one wave per token; optional gather from the sparse-T index set
__global__ __launch_bounds__(256) void rmsnorm_kernel(
    const float* __restrict__ X, const float* __restrict__ wgt,
    uint16_t* __restrict__ out, int ntok, int gather){
  int tok  = blockIdx.x * 8 + (threadIdx.x >> 5);
  int lane = threadIdx.x & 31;
  if (tok >= ntok) return;
  size_t srow;
  if (gather){
    int b_ = tok / (TP * NST);
    int i_ = (tok / NST) % TP;
    int n_ = tok % NST;
    srow = (size_t)(b_ * NST + n_) * TT + act_idx(i_);
  } else srow = (size_t)tok;
  const float* x = X + srow * DMODEL;
  float v[8]; float ss = 0.f;
  #pragma unroll
  for (int i = 0; i < 8; i++){ v[i] = x[i*32 + lane]; ss += v[i]*v[i]; }
  #pragma unroll
  for (int off = 16; off >= 1; off >>= 1) ss += __shfl_xor(ss, off);
  float r = rsqrtf(ss * (1.0f/DMODEL) + 1e-6f);
  uint16_t* o = out + (size_t)tok * DMODEL;
  #pragma unroll
  for (int i = 0; i < 8; i++) o[i*32 + lane] = f2bf(v[i] * r * wgt[i*32 + lane]);
}

// ---------------------------------------------------------------- WMMA GEMM
// C[M,N] = A[M,K](bf16, row-major) @ B^T where Bw is N x K (pre-transposed).
// 128x128 block, 8 waves, 32x64 per wave, BK = 32.
// Both tiles staged with GLOBAL_LOAD_ASYNC_TO_LDS_B128 into double buffers:
// issue tile i+1, compute tile i from LDS, s_wait_asynccnt + one barrier.
#define GF_GELU    1
#define GF_RESID   2
#define GF_SCATTER 4

__global__ __launch_bounds__(256) void gemm_bf16_kernel(
    const uint16_t* __restrict__ A, const uint16_t* __restrict__ Bw,
    const float* __restrict__ bias, uint16_t* __restrict__ outBf,
    float* __restrict__ outResid, int M, int Ntot, int K, int flags){
  __shared__ uint16_t As[2][128][40];   // [m][k], pitch 80B (16B-aligned rows)
  __shared__ uint16_t Bs[2][128][40];   // [n][k]
  const int tid  = threadIdx.x;
  const int wave = tid >> 5, lane = tid & 31;
  const int wr = wave >> 1, wc = wave & 1;     // 4x2 wave grid
  const int rowBase = blockIdx.y * 128, colBase = blockIdx.x * 128;
  FragC acc[2][4];
  #pragma unroll
  for (int a = 0; a < 2; a++)
    #pragma unroll
    for (int b = 0; b < 4; b++)
      #pragma unroll
      for (int r = 0; r < 8; r++) acc[a][b].f[r] = 0.f;

  const uint32_t asBase = lds_off_of(&As[0][0][0]);
  const uint32_t bsBase = lds_off_of(&Bs[0][0][0]);
  const uint32_t bufB   = 128u * 80u;          // bytes per buffer

  // each thread stages 2 x 16B of A and 2 x 16B of B per tile
  const int stRow = tid >> 2, stSeg = tid & 3; // 512 transfers = 128 rows x 4 segs
  auto issueTile = [&](int k0, int sel){
    #pragma unroll
    for (int j = 0; j < 2; j++){
      int row = stRow + j*64;
      uint32_t loff = (uint32_t)(sel*bufB + row*80 + stSeg*16);
      async_copy_b128(asBase + loff, A  + (size_t)(rowBase + row) * K + k0 + stSeg*8);
      async_copy_b128(bsBase + loff, Bw + (size_t)(colBase + row) * K + k0 + stSeg*8);
    }
  };

  const int nk = K >> 5;
  issueTile(0, 0);
  wait_async0();
  __syncthreads();

  for (int i = 0; i < nk; i++){
    const int cur = i & 1;
    if (i + 1 < nk) issueTile((i + 1) << 5, cur ^ 1);
    if (i + 2 < nk){   // prefetch tile i+2 into L2
      int k2 = (i + 2) << 5;
      __builtin_prefetch(A  + (size_t)(rowBase + stRow) * K + k2 + stSeg*8, 0, 1);
      __builtin_prefetch(Bw + (size_t)(colBase + stRow) * K + k2 + stSeg*8, 0, 1);
    }

    FragA af[2], bf4[4];
    const int am = lane & 15, ak = (lane >> 4) << 3;   // A: K split 0/8
    #pragma unroll
    for (int mt = 0; mt < 2; mt++){
      const uint16_t* arow = &As[cur][wr*32 + mt*16 + am][0];
      #pragma unroll
      for (int r = 0; r < 4; r++) af[mt].u[r]   = *(const uint32_t*)&arow[ak + 2*r];
      #pragma unroll
      for (int r = 0; r < 4; r++) af[mt].u[4+r] = *(const uint32_t*)&arow[16 + ak + 2*r];
    }
    const int bn = lane & 15, bk = (lane >> 4) << 3;   // B: K split 0/8
    #pragma unroll
    for (int nt = 0; nt < 4; nt++){
      const uint16_t* brow = &Bs[cur][wc*64 + nt*16 + bn][0];
      #pragma unroll
      for (int r = 0; r < 8; r++) bf4[nt].u[r] = *(const uint32_t*)&brow[bk + 2*(r & 3) + (r >> 2)*16];
    }
    #pragma unroll
    for (int mt = 0; mt < 2; mt++)
      #pragma unroll
      for (int nt = 0; nt < 4; nt++)
        acc[mt][nt].v = __builtin_amdgcn_wmma_f32_16x16x32_bf16(
            false, af[mt].v, false, bf4[nt].v, (short)0, acc[mt][nt].v, false, false);

    wait_async0();     // tile i+1 landed (overlapped with the math above)
    __syncthreads();
  }

  // epilogue: C layout lane = n%16 (+16 for m>=8), reg = m%8
  const int en = lane & 15, eh = lane >> 4;
  #pragma unroll
  for (int mt = 0; mt < 2; mt++){
    #pragma unroll
    for (int nt = 0; nt < 4; nt++){
      int gn = colBase + wc*64 + nt*16 + en;
      float bv = bias ? bias[gn] : 0.f;
      #pragma unroll
      for (int r = 0; r < 8; r++){
        int m = rowBase + wr*32 + mt*16 + r + eh*8;
        float v = acc[mt][nt].f[r] + bv;
        if (flags & GF_GELU) v = gelu_tanh(v);
        if (flags & GF_RESID){
          size_t orow;
          if (flags & GF_SCATTER){   // stock branch: scatter-add into h
            int b_ = m / (TP * NST);
            int i_ = (m / NST) % TP;
            int n_ = m % NST;
            orow = (size_t)(b_ * NST + n_) * TT + act_idx(i_);
          } else orow = (size_t)m;
          outResid[orow * Ntot + gn] += v;
        } else {
          outBf[(size_t)m * Ntot + gn] = f2bf(v);
        }
      }
    }
  }
}

// ------------------------------------------------------- fused attention
// one block per (seq, head); 4 waves, each wave owns 4 query tiles of 16.
// Q,K,V,O layout: [seq*L + t][head*32 + hd] bf16, L = 256, HDIM = 32.
__global__ __launch_bounds__(128) void attn_kernel(
    const uint16_t* __restrict__ Q, const uint16_t* __restrict__ Kb,
    const uint16_t* __restrict__ Vb, uint16_t* __restrict__ O,
    const int* __restrict__ mask, int tpPerB){
  extern __shared__ char smem[];
  uint16_t* Ks  = (uint16_t*)smem;               // [256][32] key-major
  uint16_t* Vt  = (uint16_t*)(smem + 16*1024);   // [32][256] hd-major (transposed)
  float*    Sb  = (float*)   (smem + 32*1024);   // [4][16][256] scores f32
  uint16_t* Pb  = (uint16_t*)(smem + 96*1024);   // [4][16][256] probs bf16
  float* rowInv = (float*)   (smem + 128*1024);  // [4][16]

  const int seq = blockIdx.x >> 3, head = blockIdx.x & 7;
  const int tid = threadIdx.x, wave = tid >> 5, lane = tid & 31;
  const uint32_t* Vu = (const uint32_t*)Vb;
  const uint32_t* Qu = (const uint32_t*)Q;
  uint32_t* Ksu = (uint32_t*)Ks;

  // stage K (256x32) via async B128 copies: rows are 64B = 4 transfers
  const uint32_t ksBase = lds_off_of(Ks);
  #pragma unroll
  for (int j = 0; j < 8; j++){
    int idx = j*128 + tid;                     // 1024 transfers
    int row = idx >> 2, seg = idx & 3;
    const uint16_t* g = Kb + (size_t)(seq*TT + row) * DMODEL + head*HDIM + seg*8;
    async_copy_b128(ksBase + (uint32_t)(row*64 + seg*16), g);
  }
  // stage V transposed (32x256) with sync loads (activation: runtime transpose)
  uint32_t vv[32];
  #pragma unroll
  for (int i = 0; i < 32; i++){
    int d = i*128 + tid;
    vv[i] = Vu[(size_t)(seq*TT + (d >> 4))*128 + head*16 + (d & 15)];
  }
  #pragma unroll
  for (int i = 0; i < 32; i++){
    int d = i*128 + tid;
    int key = d >> 4, hp = d & 15;
    Vt[(hp*2  )*256 + key] = (uint16_t)(vv[i] & 0xffffu);
    Vt[(hp*2+1)*256 + key] = (uint16_t)(vv[i] >> 16);
  }
  wait_async0();
  __syncthreads();

  float*    S  = Sb + wave*16*256;
  uint16_t* P  = Pb + wave*16*256;
  const uint32_t* Pu  = (const uint32_t*)P;
  const uint32_t* Vtu = (const uint32_t*)Vt;
  const int sh2 = lane >> 4;

  for (int qt = wave; qt < 16; qt += 4){
    // Q fragment (A layout): lane = query row, K split 0/8 across lane halves
    FragA qf;
    {
      const int m = lane & 15, kb = sh2 << 3;
      size_t row = (size_t)(seq*TT + qt*16 + m)*128 + head*16;
      #pragma unroll
      for (int r = 0; r < 4; r++) qf.u[r]   = Qu[row + ((kb + 2*r) >> 1)];
      #pragma unroll
      for (int r = 0; r < 4; r++) qf.u[4+r] = Qu[row + ((16 + kb + 2*r) >> 1)];
    }
    // scores: S(16x16)_j = Q(16x32) x K_j^T(32x16), one WMMA per key tile
    const int sn = lane & 15;
    #pragma unroll
    for (int j = 0; j < 16; j++){
      FragA kf;
      int key = j*16 + sn;          // B layout: lane = n (key), K split 0/8
      int kb2 = sh2 << 3;
      #pragma unroll
      for (int r = 0; r < 8; r++) kf.u[r] = Ksu[key*16 + (kb2 >> 1) + (r & 3) + (r >> 2)*8];
      FragC s;
      #pragma unroll
      for (int r = 0; r < 8; r++) s.f[r] = 0.f;
      s.v = __builtin_amdgcn_wmma_f32_16x16x32_bf16(
          false, qf.v, false, kf.v, (short)0, s.v, false, false);
      #pragma unroll
      for (int r = 0; r < 8; r++) S[(r + sh2*8)*256 + j*16 + sn] = s.f[r];
    }
    // softmax: lane handles row = lane&15, cols split 0-127 / 128-255 by half
    const int row = lane & 15, cbase = sh2 * 128;
    int bIdx = mask ? (seq / tpPerB) : 0;
    float mx = -3.0e38f;
    for (int c = 0; c < 128; c++){
      float v = S[row*256 + cbase + c] * SCALE_QK;
      if (mask && mask[bIdx*NST + cbase + c] == 0) v = -1.0e9f;
      S[row*256 + cbase + c] = v;
      mx = fmaxf(mx, v);
    }
    mx = fmaxf(mx, __shfl_xor(mx, 16));
    float sum = 0.f;
    for (int c = 0; c < 128; c++){
      float p = __expf(S[row*256 + cbase + c] - mx);
      sum += p;
      P[row*256 + cbase + c] = f2bf(p);
    }
    sum += __shfl_xor(sum, 16);
    if (lane < 16) rowInv[wave*16 + row] = 1.0f / sum;

    // O(16x32) = P(16x256) x V(256x32): 8 K-chunks x 2 n-tiles of WMMA
    FragC o0, o1;
    #pragma unroll
    for (int r = 0; r < 8; r++){ o0.f[r] = 0.f; o1.f[r] = 0.f; }
    const int pm = lane & 15, pkb = sh2 << 3;
    const int vn = lane & 15, vkb = sh2 << 3;
    for (int kk = 0; kk < 8; kk++){
      FragA pf, vf0, vf1;
      #pragma unroll
      for (int r = 0; r < 4; r++) pf.u[r]   = Pu[pm*128 + kk*16 + ((pkb + 2*r) >> 1)];
      #pragma unroll
      for (int r = 0; r < 4; r++) pf.u[4+r] = Pu[pm*128 + kk*16 + ((16 + pkb + 2*r) >> 1)];
      #pragma unroll
      for (int r = 0; r < 8; r++){
        vf0.u[r] = Vtu[(vn     )*128 + kk*16 + (vkb >> 1) + (r & 3) + (r >> 2)*8];
        vf1.u[r] = Vtu[(vn + 16)*128 + kk*16 + (vkb >> 1) + (r & 3) + (r >> 2)*8];
      }
      o0.v = __builtin_amdgcn_wmma_f32_16x16x32_bf16(
          false, pf.v, false, vf0.v, (short)0, o0.v, false, false);
      o1.v = __builtin_amdgcn_wmma_f32_16x16x32_bf16(
          false, pf.v, false, vf1.v, (short)0, o1.v, false, false);
    }
    // 1/sum scaling + store bf16
    #pragma unroll
    for (int r = 0; r < 8; r++){
      int m = r + sh2*8;
      float inv = rowInv[wave*16 + m];
      size_t orow = (size_t)(seq*TT + qt*16 + m)*DMODEL + head*HDIM;
      O[orow +      (lane & 15)] = f2bf(o0.f[r] * inv);
      O[orow + 16 + (lane & 15)] = f2bf(o1.f[r] * inv);
    }
  }
}

// ---------------------------------------------------------------- driver
extern "C" void kernel_launch(void* const* d_in, const int* in_sizes, int n_in,
                              void* d_out, int out_size, void* d_ws, size_t ws_size,
                              hipStream_t stream){
  (void)in_sizes; (void)n_in; (void)out_size; (void)ws_size;
  const float* h     = (const float*)d_in[0];
  const int*   smask = (const int*)  d_in[1];
  const float* nt_w  = (const float*)d_in[2];
  const float* ns_w  = (const float*)d_in[3];
  const float* nf_w  = (const float*)d_in[4];
  const float* Wq_t  = (const float*)d_in[5];
  const float* Wk_t  = (const float*)d_in[6];
  const float* Wv_t  = (const float*)d_in[7];
  const float* Wo_t  = (const float*)d_in[8];
  const float* Wq_s  = (const float*)d_in[9];
  const float* Wk_s  = (const float*)d_in[10];
  const float* Wv_s  = (const float*)d_in[11];
  const float* Wo_s  = (const float*)d_in[12];
  const float* W1    = (const float*)d_in[13];
  const float* b1    = (const float*)d_in[14];
  const float* W2    = (const float*)d_in[15];
  const float* b2    = (const float*)d_in[16];
  float* hout = (float*)d_out;

  char* ws = (char*)d_ws;
  uint16_t* wbuf = (uint16_t*)ws;                      // 2 MB of bf16 weights (transposed)
  uint16_t* xn = (uint16_t*)(ws + (2ull   << 20));     // 64 MB
  uint16_t* qb = (uint16_t*)(ws + (66ull  << 20));     // 64 MB
  uint16_t* kb = (uint16_t*)(ws + (130ull << 20));     // 64 MB
  uint16_t* vb = (uint16_t*)(ws + (194ull << 20));     // 64 MB
  uint16_t* ob = (uint16_t*)(ws + (258ull << 20));     // 64 MB

  uint16_t* wq_t = wbuf +      0; uint16_t* wk_t = wbuf +  65536;
  uint16_t* wv_t = wbuf + 131072; uint16_t* wo_t = wbuf + 196608;
  uint16_t* wq_s = wbuf + 262144; uint16_t* wk_s = wbuf + 327680;
  uint16_t* wv_s = wbuf + 393216; uint16_t* wo_s = wbuf + 458752;
  uint16_t* w1b  = wbuf + 524288; uint16_t* w2b  = wbuf + 786432;

  auto castw = [&](const float* s, uint16_t* d, int K, int N){
    int n = K * N;
    cast_transpose_bf16<<<(n + 255)/256, 256, 0, stream>>>(s, d, K, N);
  };
  castw(Wq_t, wq_t, DMODEL, DMODEL);  castw(Wk_t, wk_t, DMODEL, DMODEL);
  castw(Wv_t, wv_t, DMODEL, DMODEL);  castw(Wo_t, wo_t, DMODEL, DMODEL);
  castw(Wq_s, wq_s, DMODEL, DMODEL);  castw(Wk_s, wk_s, DMODEL, DMODEL);
  castw(Wv_s, wv_s, DMODEL, DMODEL);  castw(Wo_s, wo_s, DMODEL, DMODEL);
  castw(W1,   w1b,  DMODEL, FFDIM);   castw(W2,   w2b,  FFDIM, DMODEL);

  // running residual state: h -> d_out (f32)
  int n4 = S1 * DMODEL / 4;
  copy_f32v4<<<(n4 + 255)/256, 256, 0, stream>>>((const float4*)h, (float4*)hout, n4);

  const size_t attn_lds = 128*1024 + 256;

  // ---- time attention -------------------------------------------------
  rmsnorm_kernel<<<S1/8, 256, 0, stream>>>(hout, nt_w, xn, S1, 0);
  dim3 g1(DMODEL/128, S1/128);
  gemm_bf16_kernel<<<g1, 256, 0, stream>>>(xn, wq_t, nullptr, qb, nullptr, S1, DMODEL, DMODEL, 0);
  gemm_bf16_kernel<<<g1, 256, 0, stream>>>(xn, wk_t, nullptr, kb, nullptr, S1, DMODEL, DMODEL, 0);
  gemm_bf16_kernel<<<g1, 256, 0, stream>>>(xn, wv_t, nullptr, vb, nullptr, S1, DMODEL, DMODEL, 0);
  attn_kernel<<<BB*NST*HEADS, 128, attn_lds, stream>>>(qb, kb, vb, ob, nullptr, 1);
  gemm_bf16_kernel<<<g1, 256, 0, stream>>>(ob, wo_t, nullptr, nullptr, hout, S1, DMODEL, DMODEL, GF_RESID);

  // ---- stock attention (sparse-T gather / scatter) --------------------
  rmsnorm_kernel<<<S2/8, 256, 0, stream>>>(hout, ns_w, xn, S2, 1);
  dim3 g2(DMODEL/128, S2/128);
  gemm_bf16_kernel<<<g2, 256, 0, stream>>>(xn, wq_s, nullptr, qb, nullptr, S2, DMODEL, DMODEL, 0);
  gemm_bf16_kernel<<<g2, 256, 0, stream>>>(xn, wk_s, nullptr, kb, nullptr, S2, DMODEL, DMODEL, 0);
  gemm_bf16_kernel<<<g2, 256, 0, stream>>>(xn, wv_s, nullptr, vb, nullptr, S2, DMODEL, DMODEL, 0);
  attn_kernel<<<BB*TP*HEADS, 128, attn_lds, stream>>>(qb, kb, vb, ob, smask, TP);
  gemm_bf16_kernel<<<g2, 256, 0, stream>>>(ob, wo_s, nullptr, nullptr, hout, S2, DMODEL, DMODEL,
                                           GF_RESID | GF_SCATTER);

  // ---- FFN (M-chunked to bound workspace) -----------------------------
  rmsnorm_kernel<<<S1/8, 256, 0, stream>>>(hout, nf_w, xn, S1, 0);
  uint16_t* mid = qb;  // 32 MB per chunk, reuse q buffer
  const int MC = 16384;
  for (int c = 0; c < 8; c++){
    const uint16_t* xA = xn + (size_t)c * MC * DMODEL;
    dim3 ga(FFDIM/128, MC/128);
    gemm_bf16_kernel<<<ga, 256, 0, stream>>>(xA, w1b, b1, mid, nullptr, MC, FFDIM, DMODEL, GF_GELU);
    dim3 gb(DMODEL/128, MC/128);
    gemm_bf16_kernel<<<gb, 256, 0, stream>>>(mid, w2b, b2, nullptr,
                                             hout + (size_t)c * MC * DMODEL, MC, DMODEL, FFDIM,
                                             GF_RESID);
  }
}
